// PolyConv_4544075399677
// MI455X (gfx1250) — matrix operand order repaired and checked by reading, not compiled, not run on hardware
//
#include <hip/hip_runtime.h>

// ---------------------------------------------------------------------------
// PolyConv on gfx1250: h = sum_k theta[k] * adj^(k+1) @ x
// 4 chained GEMMs [8192x8192]@[8192x128], bf16-split (hi+lo) WMMA with fp32
// accumulate:  a*b ~= a_hi*b_hi + a_hi*b_lo + a_lo*b_hi  (3x v_wmma per tile)
// Software-pipelined: next tile's global loads (adj = non-temporal) issue
// before the WMMA compute phase so HBM latency hides behind the matrix ops.
// ---------------------------------------------------------------------------

typedef __attribute__((ext_vector_type(16))) __bf16 v16bf;
typedef __attribute__((ext_vector_type(8)))  __bf16 v8bf;
typedef __attribute__((ext_vector_type(4)))  __bf16 v4bf;
typedef __attribute__((ext_vector_type(8)))  float  v8f;
typedef __attribute__((ext_vector_type(4)))  float  v4f;   // clang vector (NT-load legal)

#define N_NODES 8192
#define FDIM    128
#define BM      32     // rows of adj per block
#define BN      128    // full feature width per block
#define BK      32     // K step == WMMA K for bf16
#define LPAD    8      // +8 bf16 per LDS row: 80B rows -> conflict-reduced, 16B aligned

union V16U { v16bf v; v8bf h[2]; };
union V4U  { v4bf  v; __bf16 e[4]; };

__device__ __forceinline__ void split_bf16(float x, __bf16& hi, __bf16& lo) {
    hi = (__bf16)x;
    lo = (__bf16)(x - (float)hi);
}

__global__ __launch_bounds__(256)
void polyconv_gemm_hop(const float* __restrict__ A,   // adj [N,N]
                       const float* __restrict__ X,   // feat in [N,F]
                       float* __restrict__ Y,         // feat out [N,F]
                       float* __restrict__ H,         // accum out [N,F]
                       const float* __restrict__ theta,
                       int k_idx, int first)
{
    __shared__ __align__(16) __bf16 Ah[BM][BK + LPAD];   // A hi tile
    __shared__ __align__(16) __bf16 Al[BM][BK + LPAD];   // A lo tile
    __shared__ __align__(16) __bf16 Bth[BN][BK + LPAD];  // B hi tile, K-contiguous
    __shared__ __align__(16) __bf16 Btl[BN][BK + LPAD];  // B lo tile, K-contiguous

    const int tid  = threadIdx.x;
    const int lane = tid & 31;           // wave32
    const int wave = tid >> 5;           // 8 waves
    const int wm   = wave >> 2;          // 0..1  (16-row strip of C)
    const int wn   = wave & 3;           // 0..3  (32-col strip of C)

    const int gm0 = blockIdx.x * BM;

    // staging ownership
    const int ar    = tid >> 3;          // A: row 0..31
    const int ac4   = (tid & 7) * 4;     // A: col 0,4,..,28
    const int bn    = tid & 127;         // B: output column n (0..127)
    const int bkh   = (tid >> 7) * 16;   // B: K half 0 or 16

    v8f acc[2] = {};                     // two 16x16 f32 C tiles per wave

    // ---- pipeline prologue: global loads for k0 = 0 ----
    v4f a_reg = __builtin_nontemporal_load(
        (const v4f*)(A + (size_t)(gm0 + ar) * N_NODES + ac4));
    float b_reg[16];
    #pragma unroll
    for (int j = 0; j < 16; ++j)
        b_reg[j] = X[(size_t)(bkh + j) * FDIM + bn];

    for (int k0 = 0; k0 < N_NODES; k0 += BK) {
        // ---- convert staged registers -> LDS (packed b64 stores) ----
        {
            V4U h, l;
            split_bf16(a_reg.x, h.e[0], l.e[0]);
            split_bf16(a_reg.y, h.e[1], l.e[1]);
            split_bf16(a_reg.z, h.e[2], l.e[2]);
            split_bf16(a_reg.w, h.e[3], l.e[3]);
            *(v4bf*)&Ah[ar][ac4] = h.v;
            *(v4bf*)&Al[ar][ac4] = l.v;
        }
        #pragma unroll
        for (int g = 0; g < 4; ++g) {
            V4U h, l;
            #pragma unroll
            for (int j = 0; j < 4; ++j)
                split_bf16(b_reg[g * 4 + j], h.e[j], l.e[j]);
            *(v4bf*)&Bth[bn][bkh + g * 4] = h.v;
            *(v4bf*)&Btl[bn][bkh + g * 4] = l.v;
        }
        __syncthreads();

        // ---- prefetch next K tile (overlaps the WMMA compute below) ----
        const int kn = k0 + BK;
        if (kn < N_NODES) {
            a_reg = __builtin_nontemporal_load(
                (const v4f*)(A + (size_t)(gm0 + ar) * N_NODES + kn + ac4));
            #pragma unroll
            for (int j = 0; j < 16; ++j)
                b_reg[j] = X[(size_t)(kn + bkh + j) * FDIM + bn];
        }

        // ---- A fragment (16x32 bf16, ISA A-layout: two contiguous 8-K runs) ----
        const int am   = wm * 16 + (lane & 15);
        const int koff = (lane >> 4) * 8;     // lanes 0-15: K 0..7/16..23; 16-31: 8..15/24..31
        V16U a_hi, a_lo;
        a_hi.h[0] = *(const v8bf*)&Ah[am][koff];
        a_hi.h[1] = *(const v8bf*)&Ah[am][koff + 16];
        a_lo.h[0] = *(const v8bf*)&Al[am][koff];
        a_lo.h[1] = *(const v8bf*)&Al[am][koff + 16];

        #pragma unroll
        for (int t = 0; t < 2; ++t) {
            // ---- B fragment (32x16 bf16, same K order as A: two contiguous 8-K runs) ----
            const int fn = wn * 32 + t * 16 + (lane & 15);
            const int kb = (lane >> 4) * 8;   // lanes 0-15: K 0..7/16..23; 16-31: 8..15/24..31
            V16U b_hi, b_lo;
            b_hi.h[0] = *(const v8bf*)&Bth[fn][kb];
            b_hi.h[1] = *(const v8bf*)&Bth[fn][kb + 16];
            b_lo.h[0] = *(const v8bf*)&Btl[fn][kb];
            b_lo.h[1] = *(const v8bf*)&Btl[fn][kb + 16];

            // split-precision: hi*hi + hi*lo + lo*hi
            acc[t] = __builtin_amdgcn_wmma_f32_16x16x32_bf16(
                false, a_hi.v, false, b_hi.v, (short)0, acc[t], false, false);
            acc[t] = __builtin_amdgcn_wmma_f32_16x16x32_bf16(
                false, a_hi.v, false, b_lo.v, (short)0, acc[t], false, false);
            acc[t] = __builtin_amdgcn_wmma_f32_16x16x32_bf16(
                false, a_lo.v, false, b_hi.v, (short)0, acc[t], false, false);
        }
        __syncthreads();
    }

    // ---- epilogue: Y = A@X tile; H = first ? th*Y : H + th*Y ----
    const float th   = theta[k_idx];
    const int   half = lane >> 4;
    #pragma unroll
    for (int t = 0; t < 2; ++t) {
        const int col = wn * 32 + t * 16 + (lane & 15);
        #pragma unroll
        for (int v = 0; v < 8; ++v) {
            const int row = gm0 + wm * 16 + v + 8 * half;   // C/D layout: VGPR v -> M=v(+8)
            const size_t off = (size_t)row * FDIM + col;
            const float y = acc[t][v];
            Y[off] = y;
            if (first) H[off] = th * y;
            else       H[off] = H[off] + th * y;
        }
    }
}

extern "C" void kernel_launch(void* const* d_in, const int* in_sizes, int n_in,
                              void* d_out, int out_size, void* d_ws, size_t ws_size,
                              hipStream_t stream) {
    const float* adj     = (const float*)d_in[0];
    const float* in_feat = (const float*)d_in[1];
    // d_in[2] = lapl (unused by forward)
    const float* theta   = (const float*)d_in[3];
    float* out = (float*)d_out;

    float* f0 = (float*)d_ws;                        // 8192*128 f32 = 4 MB
    float* f1 = f0 + (size_t)N_NODES * FDIM;         // second ping-pong buffer

    dim3 grid(N_NODES / BM, 1);   // 256 blocks
    dim3 block(256);              // 8 wave32 waves

    // hop 0: feat1 = adj@x;      h  = th0*feat1
    polyconv_gemm_hop<<<grid, block, 0, stream>>>(adj, in_feat, f0, out, theta, 0, 1);
    // hop 1: feat2 = adj@feat1;  h += th1*feat2
    polyconv_gemm_hop<<<grid, block, 0, stream>>>(adj, f0, f1, out, theta, 1, 0);
    // hop 2
    polyconv_gemm_hop<<<grid, block, 0, stream>>>(adj, f1, f0, out, theta, 2, 0);
    // hop 3
    polyconv_gemm_hop<<<grid, block, 0, stream>>>(adj, f0, f1, out, theta, 3, 0);
}
